// CrossAttention_23287312679508
// MI455X (gfx1250) — matrix operand order repaired and checked by reading, not compiled, not run on hardware
//
#include <hip/hip_runtime.h>
#include <hip/hip_bf16.h>

typedef _Float16 v16h __attribute__((ext_vector_type(16)));
typedef _Float16 v8h  __attribute__((ext_vector_type(8)));
typedef _Float16 v4h  __attribute__((ext_vector_type(4)));
typedef float    v8f  __attribute__((ext_vector_type(8)));

#define WMMA_F16(a, b, c) \
  __builtin_amdgcn_wmma_f32_16x16x32_f16(false, (a), false, (b), (short)0, (c), false, false)

// Async global->LDS copy of 16 bytes per lane (GLOBAL_LOAD_ASYNC_TO_LDS_B128,
// tracked by ASYNCcnt). vdst = 32-bit LDS byte offset, vaddr = 64-bit address.
__device__ __forceinline__ void async_ld_b128(const _Float16* g, _Float16* l) {
  unsigned lds = (unsigned)(unsigned long long)l;  // low 32 bits = LDS offset
  asm volatile("global_load_async_to_lds_b128 %0, %1, off"
               :: "v"(lds), "v"(g) : "memory");
}
__device__ __forceinline__ void wait_async0() {
  asm volatile("s_wait_asynccnt 0x0" ::: "memory");
}

// ---------------------------------------------------------------------------
// Cast f32 -> f16, 4-wide
// ---------------------------------------------------------------------------
__global__ __launch_bounds__(256) void ca_cast_f32_f16(const float* __restrict__ s,
                                                       _Float16* __restrict__ d, int n) {
  int i = (blockIdx.x * blockDim.x + threadIdx.x) * 4;
  if (i < n) {
    float4 v = *(const float4*)(s + i);
    v4h o;
    o[0] = (_Float16)v.x; o[1] = (_Float16)v.y;
    o[2] = (_Float16)v.z; o[3] = (_Float16)v.w;
    *(v4h*)(d + i) = o;
  }
}

// ---------------------------------------------------------------------------
// Tiled f16 WMMA GEMM: C[M x N] = A[M x K] @ B[K x N]  (f32 accumulate)
// Block tile 128x128, BK=32, 256 threads = 8 waves arranged 4(M) x 2(N),
// wave tile 32x64 = 2x4 WMMA accumulators. A staged async row-major,
// B staged transposed [n][k] so B fragments are contiguous ds_load_b128.
// Double-buffered LDS: async next tile while computing current.
// MODE 1: f16 head-major [B,H,N,64] out.  MODE 2: f32 row-major + bias.
// ---------------------------------------------------------------------------
template <int MODE>
__global__ __launch_bounds__(256) void ca_gemm_wmma(const _Float16* __restrict__ A,
                                                    const _Float16* __restrict__ Bm,
                                                    void* __restrict__ Cout,
                                                    const float* __restrict__ bias,
                                                    int M, int N, int Kd) {
  __shared__ _Float16 As[2][128 * 40];  // [m][k], k-stride 40 (32 + pad)
  __shared__ _Float16 Bt[2][128 * 40];  // [n][k], k-stride 40

  const int t    = threadIdx.x;
  const int lane = t & 31;
  const int wave = t >> 5;
  const int wm   = wave & 3;   // 4 waves along M
  const int wn   = wave >> 2;  // 2 waves along N
  const int l16  = lane & 15;
  const int hi   = lane >> 4;
  const int bm   = (int)blockIdx.y * 128;
  const int bn   = (int)blockIdx.x * 128;

  // ---- staging helpers -----------------------------------------------------
  auto stage = [&](int buf, int k0) {
    // A: 128x32 halfs = 512 xb128, async, 2 per thread
#pragma unroll
    for (int rep = 0; rep < 2; ++rep) {
      int lin = t + rep * 256;
      int row = lin >> 2;
      int c8  = (lin & 3) << 3;
      async_ld_b128(&A[(size_t)(bm + row) * Kd + k0 + c8], &As[buf][row * 40 + c8]);
    }
    // B: 32x128 halfs, transpose into Bt[n][k]; 2 x (b128 load + 8 b16 stores)
#pragma unroll
    for (int rep = 0; rep < 2; ++rep) {
      int lin  = t + rep * 256;
      int krow = lin >> 4;         // 0..31
      int c8   = (lin & 15) << 3;  // 0..120
      v8h u = *(const v8h*)&Bm[(size_t)(k0 + krow) * N + bn + c8];
#pragma unroll
      for (int j = 0; j < 8; ++j) Bt[buf][(c8 + j) * 40 + krow] = u[j];
    }
  };

  v8f acc[2][4] = {};

  stage(0, 0);
  int cur = 0;
  for (int k0 = 0; k0 < Kd; k0 += 32) {
    wait_async0();
    __syncthreads();  // publish buf[cur]; all reads of buf[cur^1] done
    if (k0 + 32 < Kd) stage(cur ^ 1, k0 + 32);
    if (k0 + 64 < Kd) {  // L2 prefetch two steps ahead (global_prefetch_b8)
      __builtin_prefetch(&A[(size_t)(bm + (t >> 1)) * Kd + k0 + 64], 0, 1);
      __builtin_prefetch(&Bm[(size_t)(k0 + 64 + (t >> 3)) * N + bn], 0, 1);
    }

    // fragments from LDS (two contiguous 8-half k runs per lane -> ds_load_b128 pairs)
    v16h af[2], bf[4];
#pragma unroll
    for (int mi = 0; mi < 2; ++mi) {
      int m = wm * 32 + mi * 16 + l16;
#pragma unroll
      for (int i = 0; i < 8; ++i) {
        int kk = ((i < 4) ? 2 * i : 16 + 2 * (i - 4)) + hi * 8;
        af[mi][2 * i]     = As[cur][m * 40 + kk];
        af[mi][2 * i + 1] = As[cur][m * 40 + kk + 1];
      }
    }
#pragma unroll
    for (int ni = 0; ni < 4; ++ni) {
      int n = wn * 64 + ni * 16 + l16;
#pragma unroll
      for (int j = 0; j < 8; ++j) {
        int kk = ((j < 4) ? 2 * j : 16 + 2 * (j - 4)) + hi * 8;
        bf[ni][2 * j]     = Bt[cur][n * 40 + kk];
        bf[ni][2 * j + 1] = Bt[cur][n * 40 + kk + 1];
      }
    }
#pragma unroll
    for (int mi = 0; mi < 2; ++mi)
#pragma unroll
      for (int ni = 0; ni < 4; ++ni)
        acc[mi][ni] = WMMA_F16(af[mi], bf[ni], acc[mi][ni]);
    cur ^= 1;
  }

  // Store: C fragment element (M = r + 8*hi, N = l16) per 16x16 tile
#pragma unroll
  for (int mi = 0; mi < 2; ++mi)
#pragma unroll
    for (int ni = 0; ni < 4; ++ni)
#pragma unroll
      for (int r = 0; r < 8; ++r) {
        int grow = bm + wm * 32 + mi * 16 + r + hi * 8;
        int gcol = bn + wn * 64 + ni * 16 + l16;
        float v  = acc[mi][ni][r];
        if (MODE == 1) {
          int b = grow >> 10, nq = grow & 1023;
          int h = gcol >> 6, d = gcol & 63;
          ((_Float16*)Cout)[(((size_t)(b * 16 + h)) * 1024 + nq) * 64 + d] =
              (_Float16)v;
        } else {
          ((float*)Cout)[(size_t)grow * N + gcol] = v + bias[gcol];
        }
      }
}

// ---------------------------------------------------------------------------
// Fused flash attention (per (b,h)): softmax(Q K^T / 8) V
// Block = 8 waves owns 128 query rows of one (b,h); wave owns 16 rows.
// K chunk staged async row-major; V chunk staged transposed [d][key] so
// the P@V B-fragments are contiguous. Double-buffered chunks of 32 keys.
// ---------------------------------------------------------------------------
__global__ __launch_bounds__(256) void ca_attn_wmma(const _Float16* __restrict__ Qh,
                                                    const _Float16* __restrict__ Kh,
                                                    const _Float16* __restrict__ Vh,
                                                    _Float16* __restrict__ AO) {
  __shared__ _Float16 Kc[2][32 * 72];   // [key][d], d-stride 72
  __shared__ _Float16 Vt[2][64 * 40];   // [d][key], key-stride 40
  __shared__ _Float16 Pl[8 * 16 * 40];  // per-wave 16x32 P patch

  const int t     = threadIdx.x;
  const int lane  = t & 31;
  const int wave  = t >> 5;
  const int l16   = lane & 15;
  const int hi    = lane >> 4;
  const int bh    = blockIdx.x >> 3;
  const int qtile = blockIdx.x & 7;
  const int qbase = qtile * 128 + wave * 16;
  const int b     = bh >> 4;
  const int h     = bh & 15;

  auto stage = [&](int buf, int kc) {
    int row = t >> 3;          // key 0..31
    int c8  = (t & 7) << 3;    // d 0..56
    size_t g = ((size_t)bh * 1024 + kc + row) * 64 + c8;
    async_ld_b128(&Kh[g], &Kc[buf][row * 72 + c8]);
    v8h u = *(const v8h*)&Vh[g];
#pragma unroll
    for (int j = 0; j < 8; ++j) Vt[buf][(c8 + j) * 40 + row] = u[j];
  };

  // Q fragments (16 rows x 64 dims -> 2 A-frags of 16x32) from global
  v16h qf[2];
  const _Float16* Qrow = Qh + ((size_t)bh * 1024 + qbase + l16) * 64;
#pragma unroll
  for (int kk2 = 0; kk2 < 2; ++kk2)
#pragma unroll
    for (int i = 0; i < 8; ++i) {
      int kq = kk2 * 32 + ((i < 4) ? 2 * i : 16 + 2 * (i - 4)) + hi * 8;
      qf[kk2][2 * i]     = Qrow[kq];
      qf[kk2][2 * i + 1] = Qrow[kq + 1];
    }

  float mrow[8], lrow[8];
  v8f Of[4] = {};
#pragma unroll
  for (int r = 0; r < 8; ++r) { mrow[r] = -1e30f; lrow[r] = 0.0f; }

  _Float16* Pw = &Pl[wave * 16 * 40];

  stage(0, 0);
  int cur = 0;
  for (int kc = 0; kc < 1024; kc += 32) {
    wait_async0();
    __syncthreads();
    if (kc + 32 < 1024) stage(cur ^ 1, kc + 32);

    // S(16x32) = Q(16x64) @ Kchunk^T : 2 n-halves x 2 k-steps
    v8f S[2] = {};
#pragma unroll
    for (int nh = 0; nh < 2; ++nh) {
      int key = nh * 16 + l16;
#pragma unroll
      for (int kk2 = 0; kk2 < 2; ++kk2) {
        v16h bf;
#pragma unroll
        for (int j = 0; j < 8; ++j) {
          int kk = kk2 * 32 + ((j < 4) ? 2 * j : 16 + 2 * (j - 4)) + hi * 8;
          bf[2 * j]     = Kc[cur][key * 72 + kk];
          bf[2 * j + 1] = Kc[cur][key * 72 + kk + 1];
        }
        S[nh] = WMMA_F16(qf[kk2], bf, S[nh]);
      }
    }

    // Online softmax (row M = r + 8*hi lives in one 16-lane half-wave)
#pragma unroll
    for (int r = 0; r < 8; ++r) {
      float s0 = S[0][r] * 0.125f;
      float s1 = S[1][r] * 0.125f;
      float mx = fmaxf(s0, s1);
      mx = fmaxf(mx, __shfl_xor(mx, 1, 16));
      mx = fmaxf(mx, __shfl_xor(mx, 2, 16));
      mx = fmaxf(mx, __shfl_xor(mx, 4, 16));
      mx = fmaxf(mx, __shfl_xor(mx, 8, 16));
      float mnew  = fmaxf(mrow[r], mx);
      float alpha = __expf(mrow[r] - mnew);
      float p0 = __expf(s0 - mnew);
      float p1 = __expf(s1 - mnew);
      float rs = p0 + p1;
      rs += __shfl_xor(rs, 1, 16);
      rs += __shfl_xor(rs, 2, 16);
      rs += __shfl_xor(rs, 4, 16);
      rs += __shfl_xor(rs, 8, 16);
      lrow[r] = lrow[r] * alpha + rs;
      mrow[r] = mnew;
      int Mr = r + hi * 8;
      Pw[Mr * 40 + l16]      = (_Float16)p0;
      Pw[Mr * 40 + 16 + l16] = (_Float16)p1;
#pragma unroll
      for (int f = 0; f < 4; ++f) Of[f][r] *= alpha;
    }
    asm volatile("s_wait_dscnt 0x0" ::: "memory");  // P visible wave-wide

    // Re-swizzle P (16x32) into A-fragment layout
    v16h pf;
#pragma unroll
    for (int i = 0; i < 8; ++i) {
      int kk = ((i < 4) ? 2 * i : 16 + 2 * (i - 4)) + hi * 8;
      pf[2 * i]     = Pw[l16 * 40 + kk];
      pf[2 * i + 1] = Pw[l16 * 40 + kk + 1];
    }

    // O(16x64) += P(16x32) @ Vchunk : 4 d-blocks, contiguous Vt fragments
#pragma unroll
    for (int f = 0; f < 4; ++f) {
      v16h bf;
#pragma unroll
      for (int j = 0; j < 8; ++j) {
        int kk = ((j < 4) ? 2 * j : 16 + 2 * (j - 4)) + hi * 8;
        bf[2 * j]     = Vt[cur][(f * 16 + l16) * 40 + kk];
        bf[2 * j + 1] = Vt[cur][(f * 16 + l16) * 40 + kk + 1];
      }
      Of[f] = WMMA_F16(pf, bf, Of[f]);
    }
    cur ^= 1;
  }

  // Normalize and store to [B, Nq, H*64] (f16) for the output projection
#pragma unroll
  for (int r = 0; r < 8; ++r) {
    float inv = 1.0f / lrow[r];
    int qrow  = qbase + r + hi * 8;
#pragma unroll
    for (int f = 0; f < 4; ++f) {
      AO[((size_t)(b * 1024 + qrow)) * 1024 + h * 64 + f * 16 + l16] =
          (_Float16)(Of[f][r] * inv);
    }
  }
}

// ---------------------------------------------------------------------------
// Host launch
// ---------------------------------------------------------------------------
extern "C" void kernel_launch(void* const* d_in, const int* in_sizes, int n_in,
                              void* d_out, int out_size, void* d_ws, size_t ws_size,
                              hipStream_t stream) {
  const float* x  = (const float*)d_in[0];
  const float* cx = (const float*)d_in[1];
  const float* Wq = (const float*)d_in[2];
  const float* Wk = (const float*)d_in[3];
  const float* Wv = (const float*)d_in[4];
  const float* Wo = (const float*)d_in[5];
  const float* bo = (const float*)d_in[6];

  const int NX = 4 * 1024 * 1024;
  const int NW = 1024 * 1024;

  _Float16* ws  = (_Float16*)d_ws;
  _Float16* xh  = ws;
  _Float16* ch  = xh + NX;
  _Float16* wqh = ch + NX;
  _Float16* wkh = wqh + NW;
  _Float16* wvh = wkh + NW;
  _Float16* woh = wvh + NW;
  _Float16* Qf  = woh + NW;  // head-major [B,H,Nq,64]
  _Float16* Kf  = Qf + NX;
  _Float16* Vf  = Kf + NX;
  _Float16* AOf = Vf + NX;   // [B,Nq,1024]

  ca_cast_f32_f16<<<(NX / 4 + 255) / 256, 256, 0, stream>>>(x, xh, NX);
  ca_cast_f32_f16<<<(NX / 4 + 255) / 256, 256, 0, stream>>>(cx, ch, NX);
  ca_cast_f32_f16<<<(NW / 4 + 255) / 256, 256, 0, stream>>>(Wq, wqh, NW);
  ca_cast_f32_f16<<<(NW / 4 + 255) / 256, 256, 0, stream>>>(Wk, wkh, NW);
  ca_cast_f32_f16<<<(NW / 4 + 255) / 256, 256, 0, stream>>>(Wv, wvh, NW);
  ca_cast_f32_f16<<<(NW / 4 + 255) / 256, 256, 0, stream>>>(Wo, woh, NW);

  dim3 ggrid(1024 / 128, 4096 / 128);
  ca_gemm_wmma<1><<<ggrid, 256, 0, stream>>>(xh, wqh, Qf, nullptr, 4096, 1024, 1024);
  ca_gemm_wmma<1><<<ggrid, 256, 0, stream>>>(ch, wkh, Kf, nullptr, 4096, 1024, 1024);
  ca_gemm_wmma<1><<<ggrid, 256, 0, stream>>>(ch, wvh, Vf, nullptr, 4096, 1024, 1024);

  ca_attn_wmma<<<512, 256, 0, stream>>>(Qf, Kf, Vf, AOf);

  ca_gemm_wmma<2><<<ggrid, 256, 0, stream>>>(AOf, woh, (float*)d_out, bo,
                                             4096, 1024, 1024);
}
